// RNN_28690381537694
// MI455X (gfx1250) — matrix operand order repaired and checked by reading, not compiled, not run on hardware
//
#include <hip/hip_runtime.h>
#include <math.h>

// ---- Problem constants (match reference: B=2048, T=1024, I=1, H=64) ----
#define RNN_B 2048
#define RNN_T 1024
#define RNN_H 64

typedef _Float16 v16h  __attribute__((ext_vector_type(16)));
typedef __fp16   pkh2  __attribute__((ext_vector_type(2)));  // cvt_pkrtz result type
typedef float    v8f   __attribute__((ext_vector_type(8)));
typedef int      i8v   __attribute__((ext_vector_type(8)));

#if defined(__has_builtin)
#if __has_builtin(__builtin_amdgcn_tanhf)
#define RNN_TANHF(x) __builtin_amdgcn_tanhf(x)
#else
#define RNN_TANHF(x) tanhf(x)
#endif
#else
#define RNN_TANHF(x) tanhf(x)
#endif

// Exchange a dword with the XOR-16 partner lane (wave32): ds_swizzle SWAPX16.
// BITMODE encoding: offset = (xor<<10)|(or<<5)|and = (0x10<<10)|0x1f = 0x401f.
__device__ __forceinline__ int swap16(int v) {
    return __builtin_amdgcn_ds_swizzle(v, 0x401f);
}
__device__ __forceinline__ float swap16f(float v) {
    return __builtin_bit_cast(float, swap16(__builtin_bit_cast(int, v)));
}
// Pack two f32 -> packed f16 pair (low half = a).
__device__ __forceinline__ int pack2(float a, float b) {
    pkh2 p = __builtin_amdgcn_cvt_pkrtz(a, b);
    return __builtin_bit_cast(int, p);
}

// One wave (32 threads) owns a 16-row batch tile and runs the full T-step
// recurrence.  Orientation: D[n][b] = W_hh (A: 16n x 64k) * h^T (B: 64k x 16b).
// W_hh lives in registers as 8 pre-packed f16 A-fragments for the whole kernel.
__global__ __launch_bounds__(32)
void rnn_wmma_kernel(const float* __restrict__ x,      // [B, T, 1]
                     const float* __restrict__ W_ih,   // [H, 1]
                     const float* __restrict__ W_hh,   // [H, H]
                     const float* __restrict__ b_ih,   // [H]
                     const float* __restrict__ b_hh,   // [H]
                     const float* __restrict__ W_out,  // [1, H]
                     const float* __restrict__ b_out,  // [1]
                     float* __restrict__ out)          // [B, 1]
{
    const int lane = threadIdx.x;     // 0..31
    const int l16  = lane & 15;       // batch column within tile
    const int g    = lane >> 4;       // lane group 0/1
    const int tile = blockIdx.x;      // 0..B/16-1
    const int brow = tile * 16 + l16; // this lane's batch row

    // ---- W_hh as resident A-fragments: Wf[ntile][kchunk], f16-packed ----
    // A 16x32 f16 fragment, contiguous k order used here (B matches): lane M = l16; VGPR v holds K pair starting at
    // bk = g*16 + v*2 within the 32-wide k-chunk.
    i8v Wf[4][2];
#pragma unroll
    for (int nt = 0; nt < 4; ++nt) {
        const int row = nt * 16 + l16;
#pragma unroll
        for (int kc = 0; kc < 2; ++kc) {
#pragma unroll
            for (int v = 0; v < 8; ++v) {
                const int bk = g * 16 + v * 2;
                const int k  = kc * 32 + bk;
                const float a0 = W_hh[row * RNN_H + k];
                const float a1 = W_hh[row * RNN_H + k + 1];
                Wf[nt][kc][v] = pack2(a0, a1);
            }
        }
    }

    // ---- Per-lane elementwise constants in D layout: n = nt*16 + 8*g + v ----
    float wih[4][8], bias[4][8];
#pragma unroll
    for (int nt = 0; nt < 4; ++nt)
#pragma unroll
        for (int v = 0; v < 8; ++v) {
            const int n = nt * 16 + 8 * g + v;
            wih[nt][v]  = W_ih[n];                 // I == 1
            bias[nt][v] = b_ih[n] + b_hh[n];
        }

    // ---- Recurrence ----
    i8v Bk0 = {0, 0, 0, 0, 0, 0, 0, 0};  // h^T fragment, K = 0..31  (h0 = 0)
    i8v Bk1 = {0, 0, 0, 0, 0, 0, 0, 0};  // h^T fragment, K = 32..63
    float e[4][8];                        // h in f32, D layout

    const float* __restrict__ xr = x + (size_t)brow * RNN_T;
    float xcur = xr[0];

    for (int t = 0; t < RNN_T; ++t) {
        // prefetch next x (clamped so the load is unconditional & in-bounds)
        const int tn = (t + 1 < RNN_T) ? (t + 1) : (RNN_T - 1);
        const float xnxt = xr[tn];

        const v16h B0 = __builtin_bit_cast(v16h, Bk0);
        const v16h B1 = __builtin_bit_cast(v16h, Bk1);

#pragma unroll
        for (int nt = 0; nt < 4; ++nt) {
            v8f c = {};
            c = __builtin_amdgcn_wmma_f32_16x16x32_f16(
                    false, __builtin_bit_cast(v16h, Wf[nt][0]), false, B0,
                    (short)0, c, false, false);
            c = __builtin_amdgcn_wmma_f32_16x16x32_f16(
                    false, __builtin_bit_cast(v16h, Wf[nt][1]), false, B1,
                    (short)0, c, false, false);
#pragma unroll
            for (int v = 0; v < 8; ++v)
                e[nt][v] = RNN_TANHF(c[v] + xcur * wih[nt][v] + bias[nt][v]);
        }

        // ---- D -> next B relayout: pack pairs, XOR-16 exchange, select ----
        int pk[4][4], sx[4][4];
#pragma unroll
        for (int nt = 0; nt < 4; ++nt)
#pragma unroll
            for (int j = 0; j < 4; ++j)
                pk[nt][j] = pack2(e[nt][2 * j], e[nt][2 * j + 1]);
#pragma unroll
        for (int nt = 0; nt < 4; ++nt)
#pragma unroll
            for (int j = 0; j < 4; ++j)
                sx[nt][j] = swap16(pk[nt][j]);

#pragma unroll
        for (int j = 0; j < 4; ++j) {
            // B0: group0 needs k=0..15 (nt0), group1 needs k=16..31 (nt1)
            Bk0[j]     = (g == 0) ? pk[0][j] : sx[1][j];
            Bk0[4 + j] = (g == 0) ? sx[0][j] : pk[1][j];
            // B1: group0 needs k=32..47 (nt2), group1 needs k=48..63 (nt3)
            Bk1[j]     = (g == 0) ? pk[2][j] : sx[3][j];
            Bk1[4 + j] = (g == 0) ? sx[2][j] : pk[3][j];
        }

        xcur = xnxt;
    }

    // ---- Output head: s = h_T . W_out, XOR-16 reduce, sigmoid ----
    float s = 0.0f;
#pragma unroll
    for (int nt = 0; nt < 4; ++nt)
#pragma unroll
        for (int v = 0; v < 8; ++v)
            s += e[nt][v] * W_out[nt * 16 + 8 * g + v];
    s += swap16f(s);

    const float z = s + b_out[0];
    const float r = 1.0f / (1.0f + __expf(-z));
    if (lane < 16)
        out[tile * 16 + l16] = r;
}

extern "C" void kernel_launch(void* const* d_in, const int* in_sizes, int n_in,
                              void* d_out, int out_size, void* d_ws, size_t ws_size,
                              hipStream_t stream) {
    (void)in_sizes; (void)n_in; (void)out_size; (void)d_ws; (void)ws_size;
    const float* x     = (const float*)d_in[0];
    const float* W_ih  = (const float*)d_in[1];
    const float* W_hh  = (const float*)d_in[2];
    const float* b_ih  = (const float*)d_in[3];
    const float* b_hh  = (const float*)d_in[4];
    const float* W_out = (const float*)d_in[5];
    const float* b_out = (const float*)d_in[6];
    float* out = (float*)d_out;

    dim3 grid(RNN_B / 16);   // 128 independent 16-row tiles, one wave each
    dim3 block(32);          // exactly one wave32: EXEC all-ones for WMMA
    rnn_wmma_kernel<<<grid, block, 0, stream>>>(x, W_ih, W_hh, b_ih, b_hh,
                                                W_out, b_out, out);
}
